// IWTBlock_17952963298083
// MI455X (gfx1250) — matrix operand order repaired and checked by reading, not compiled, not run on hardware
//
#include <hip/hip_runtime.h>
#include <stdint.h>

// MI455X / gfx1250, wave32.
// 1x1-conv GEMM (M=256,K=256,N=262144) in bf16 via v_wmma_f32_16x16x32_bf16,
// weights pre-converted to fragment-ordered bf16 (d_ws) and staged to LDS with
// CDNA5 async-to-LDS copies; fused bias + inverse-Haar butterfly epilogue.

typedef __bf16 bf16_t;
typedef bf16_t v16bf __attribute__((ext_vector_type(16)));
typedef float  v8f   __attribute__((ext_vector_type(8)));

// Hardware packed fp32->bf16 (RNE), one VALU op.
__device__ __forceinline__ uint32_t pk_bf16(float lo, float hi) {
  uint32_t r;
  asm("v_cvt_pk_bf16_f32 %0, %1, %2" : "=v"(r) : "v"(lo), "v"(hi));
  return r;
}

// CDNA5 async copy: 16 bytes per lane, global -> LDS, tracked by ASYNCcnt.
__device__ __forceinline__ void async_cp16(uint32_t lds_off, const uint32_t* gsrc) {
  uint64_t ga = (uint64_t)(uintptr_t)gsrc;
  asm volatile("global_load_async_to_lds_b128 %0, %1, off"
               :: "v"(lds_off), "v"(ga) : "memory");
}
__device__ __forceinline__ void wait_async0() {
  asm volatile("s_wait_asynccnt 0x0" ::: "memory");
}

// ---------------------------------------------------------------------------
// Pre-pass: conv_w fp32 [256x256] -> bf16 packed in WMMA A-fragment order.
// Layout: dword[kt*4096 + (mt*32+ln)*8 + v], kt=K-tile(32), mt=M-tile(16).
// A-matrix 16x32 bf16: lanes 0-15 = row M, K=0..15 in v0..7 (packed pairs);
// lanes 16-31 same rows, K=16..31 (matches B fragment K order).
// ---------------------------------------------------------------------------
__global__ void convert_w_kernel(const float* __restrict__ w,
                                 uint32_t* __restrict__ ws) {
  const int idx = blockIdx.x * 256 + threadIdx.x;  // 0..32767
  const int v   = idx & 7;
  const int e   = (idx >> 3) & 511;
  const int kt  = idx >> 12;
  const int ln  = e & 31, mt = e >> 5;
  const int m   = mt * 16 + (ln & 15);
  const int kb  = kt * 32 + ((ln >> 4) << 4) + (v << 1);
  const float* s = w + m * 256 + kb;
  ws[idx] = pk_bf16(s[0], s[1]);
}

// Stage one K-step of pre-converted weights (16 KB) via async-to-LDS.
// Thread t copies 64 contiguous bytes (4 x b128 lanes-worth).
__device__ __forceinline__ void fillW_async(uint32_t lds_base,
                                            const uint32_t* __restrict__ wsrc_kt,
                                            int t) {
  const uint32_t* g = wsrc_kt + t * 16;   // dwords
  const uint32_t  l = lds_base + t * 64;  // bytes
  async_cp16(l + 0,  g + 0);
  async_cp16(l + 16, g + 4);
  async_cp16(l + 32, g + 8);
  async_cp16(l + 48, g + 12);
}

// Stage one K-step of x (32 ch x 128 px) as lx[n][kp] = packed
// (K=kc0+2kp, K=kc0+2kp+1) at pixel n. B-fragment (32x16 bf16): lane 0-15 =
// column n, K=0..15 in v0..v7; lanes 16-31: K=16..31 -> one 32B read per lane.
__device__ __forceinline__ void fillX(uint32_t* __restrict__ dst,
                                      const float* __restrict__ xb,
                                      int kc0, int t) {
  const int kp = t >> 4;                     // 0..15 channel pair
  const int n0 = (t & 15) << 3;              // 0..120 pixel run
  const float* s0 = xb + (size_t)(kc0 + 2 * kp) * 16384 + n0;
  const float* s1 = s0 + 16384;              // next channel (H*W stride)
  float4 a0 = *(const float4*)(s0);
  float4 a1 = *(const float4*)(s0 + 4);
  float4 c0 = *(const float4*)(s1);
  float4 c1 = *(const float4*)(s1 + 4);
  uint32_t* d = dst + n0 * 16 + kp;
  d[0 * 16] = pk_bf16(a0.x, c0.x); d[1 * 16] = pk_bf16(a0.y, c0.y);
  d[2 * 16] = pk_bf16(a0.z, c0.z); d[3 * 16] = pk_bf16(a0.w, c0.w);
  d[4 * 16] = pk_bf16(a1.x, c1.x); d[5 * 16] = pk_bf16(a1.y, c1.y);
  d[6 * 16] = pk_bf16(a1.z, c1.z); d[7 * 16] = pk_bf16(a1.w, c1.w);
}

__launch_bounds__(256, 1)
__global__ void iwt_conv_kernel(const float* __restrict__ x,
                                const uint32_t* __restrict__ wbf,  // d_ws
                                const float* __restrict__ bias,
                                float* __restrict__ out) {
  __shared__ __align__(32) uint32_t lw[2][16 * 32 * 8]; // 2 x 16 KB weight frags
  __shared__ __align__(32) uint32_t lx[2][128 * 16];    // 2 x 8 KB x frags
  __shared__ __align__(16) float lbias2[256];           // Haar-combined bias

  const int t    = threadIdx.x;
  const int lane = t & 31;
  const int wave = t >> 5;      // 8 waves
  const int wm   = wave >> 1;   // 0..3 -> M block of 64
  const int wn   = wave & 1;    // 0..1 -> N block of 64

  const int b = blockIdx.x >> 7;   // batch
  const int h = blockIdx.x & 127;  // input row

  const float* xb = x + (size_t)b * 256 * 16384 + (size_t)h * 128;

  // Haar-combined bias: B[i][j] = 0.5*(b0 + si*b1 + sj*b2 + si*sj*b3),
  // si = i?-1:+1 (row), sj = j?-1:+1 (col); k = t&3 = {00,01,10,11}.
  {
    const int g = t >> 2, k = t & 3;
    const float* bp = bias + g * 4;
    const float si = (k & 2) ? -1.f : 1.f;
    const float sj = (k & 1) ? -1.f : 1.f;
    lbias2[t] = 0.5f * (bp[0] + si * bp[1] + sj * bp[2] + si * sj * bp[3]);
  }

  const uint32_t lw_off0 = (uint32_t)(uintptr_t)&lw[0][0];
  const uint32_t lw_off1 = (uint32_t)(uintptr_t)&lw[1][0];

  v8f acc[4][4];
#pragma unroll
  for (int i = 0; i < 4; ++i)
#pragma unroll
    for (int j = 0; j < 4; ++j)
      acc[i][j] = (v8f){0.f, 0.f, 0.f, 0.f, 0.f, 0.f, 0.f, 0.f};

  fillW_async(lw_off0, wbf, t);
  fillX(&lx[0][0], xb, 0, t);
  wait_async0();
  __syncthreads();

  for (int ks = 0; ks < 8; ++ks) {           // K = 256 in steps of 32
    const uint32_t* curW = &lw[ks & 1][0];
    const uint32_t* curX = &lx[ks & 1][0];
    if (ks + 1 < 8) {                        // prefetch next tiles (other buffer)
      fillW_async((ks & 1) ? lw_off0 : lw_off1, wbf + (ks + 1) * 4096, t);
      fillX(&lx[(ks + 1) & 1][0], xb, (ks + 1) * 32, t);
    }

    v16bf afr[4], bfr[4];
#pragma unroll
    for (int mti = 0; mti < 4; ++mti) {
      const int tile = wm * 4 + mti;
      afr[mti] = *reinterpret_cast<const v16bf*>(curW + (tile * 32 + lane) * 8);
    }
#pragma unroll
    for (int nti = 0; nti < 4; ++nti) {
      const int n = wn * 64 + nti * 16 + (lane & 15);
      bfr[nti] = *reinterpret_cast<const v16bf*>(curX + n * 16 + ((lane >> 4) << 3));
    }

#pragma unroll
    for (int mti = 0; mti < 4; ++mti)
#pragma unroll
      for (int nti = 0; nti < 4; ++nti)
        acc[mti][nti] = __builtin_amdgcn_wmma_f32_16x16x32_bf16(
            false, afr[mti], false, bfr[nti], (short)0, acc[mti][nti],
            false, false);

    wait_async0();                           // our async LDS writes are done
    __syncthreads();                         // everyone's are visible
  }

  // Fused epilogue: inverse Haar butterfly (taps +-0.5) + folded bias.
  // C/D layout: M = Mtile + vgpr + 8*(lane>=16), N = Ntile + (lane&15);
  // a Haar group = 4 consecutive acc elements of one lane.
  const int halfs = lane >> 4;
  const int nlo   = lane & 15;
  float* outb = out + (size_t)b * 64 * 65536;   // 64 ch * 256*256
  const int row0 = 2 * h;
#pragma unroll
  for (int mti = 0; mti < 4; ++mti) {
#pragma unroll
    for (int rb = 0; rb < 2; ++rb) {
      const int M0 = wm * 64 + mti * 16 + halfs * 8 + rb * 4;
      const int g  = M0 >> 2;                   // output channel
      const float4 Bc = *reinterpret_cast<const float4*>(&lbias2[g * 4]);
#pragma unroll
      for (int nti = 0; nti < 4; ++nti) {
        const int wp = wn * 64 + nti * 16 + nlo;
        const float a0 = acc[mti][nti][rb * 4 + 0];
        const float a1 = acc[mti][nti][rb * 4 + 1];
        const float a2 = acc[mti][nti][rb * 4 + 2];
        const float a3 = acc[mti][nti][rb * 4 + 3];
        const float u = a0 + a1, v = a2 + a3;
        const float p = a0 - a1, q = a2 - a3;
        const float s00 = fmaf(0.5f, u + v, Bc.x);  // (2h,   2w)
        const float s01 = fmaf(0.5f, u - v, Bc.y);  // (2h,   2w+1)
        const float s10 = fmaf(0.5f, p + q, Bc.z);  // (2h+1, 2w)
        const float s11 = fmaf(0.5f, p - q, Bc.w);  // (2h+1, 2w+1)
        float* pp = outb + (size_t)g * 65536 + (size_t)row0 * 256 + 2 * wp;
        *reinterpret_cast<float2*>(pp)       = make_float2(s00, s01);
        *reinterpret_cast<float2*>(pp + 256) = make_float2(s10, s11);
      }
    }
  }
}

extern "C" void kernel_launch(void* const* d_in, const int* in_sizes, int n_in,
                              void* d_out, int out_size, void* d_ws, size_t ws_size,
                              hipStream_t stream) {
  const float* x    = (const float*)d_in[0];  // [16,256,128,128]
  const float* w    = (const float*)d_in[1];  // [256,256]
  const float* bias = (const float*)d_in[2];  // [256]
  float* out        = (float*)d_out;          // [16,64,256,256]
  uint32_t* wbf     = (uint32_t*)d_ws;        // 128 KB: fragment-ordered bf16 W

  // One-time (per launch) weight convert+reorder, then the fused GEMM+IWT.
  convert_w_kernel<<<dim3(128), dim3(256), 0, stream>>>(w, wbf);
  iwt_conv_kernel<<<dim3(16 * 128), dim3(256), 0, stream>>>(x, wbf, bias, out);
}
